// QKVAttention_89799176225591
// MI455X (gfx1250) — matrix operand order repaired and checked by reading, not compiled, not run on hardware
//
#include <hip/hip_runtime.h>
#include <math.h>

typedef __attribute__((ext_vector_type(16))) _Float16 v16h;
typedef __attribute__((ext_vector_type(8)))  _Float16 v8h;
typedef __attribute__((ext_vector_type(8)))  float    v8f;
typedef __attribute__((ext_vector_type(4)))  float    v4f;

#define DIM     192
#define TSEQ    4096
#define NBATCH  8
#define BM      128      // query rows per workgroup (8 waves x 16)
#define BN      32       // keys per S-tile
#define NT      (TSEQ / BN)
#define NWAVES  8
#define CT      12       // DIM / 16 output column tiles
#define KSTEPS  6        // DIM / 32 k-steps for QK^T
#define SCALE_F 0.07216878364870322f   // 192^-0.5

// load next K/V tile rows into registers (global -> VGPR, latency hidden by WMMA)
__device__ __forceinline__ void load_tile_regs(const float* base, v4f* kreg, v4f* vreg)
{
#pragma unroll
    for (int j = 0; j < 6; ++j) kreg[j] = *(const v4f*)(base + 4 * j);
#pragma unroll
    for (int j = 0; j < 6; ++j) vreg[j] = *(const v4f*)(base + DIM + 4 * j);
}

// convert + store a staged tile into one LDS buffer (incl. ||k||^2 reduction)
__device__ __forceinline__ void store_tile_lds(_Float16* sKp, _Float16* sVtp,
                                               float* sK2p, const v4f* kreg,
                                               const v4f* vreg, int kid, int cp0,
                                               int tid)
{
    float k2p = 0.f;
    v8h kh[3];
#pragma unroll
    for (int j = 0; j < 6; ++j)
#pragma unroll
        for (int i = 0; i < 4; ++i) {
            float s = kreg[j][i] * SCALE_F;
            k2p += s * s;
            kh[j >> 1][(j & 1) * 4 + i] = (_Float16)s;
        }
    _Float16* kd = sKp + kid * DIM + cp0;
#pragma unroll
    for (int j = 0; j < 3; ++j) *(v8h*)(kd + 8 * j) = kh[j];

    k2p += __shfl_xor(k2p, 1, 32);
    k2p += __shfl_xor(k2p, 2, 32);
    k2p += __shfl_xor(k2p, 4, 32);
    if ((tid & 7) == 0) sK2p[kid] = k2p;

#pragma unroll
    for (int j = 0; j < 6; ++j)
#pragma unroll
        for (int i = 0; i < 4; ++i)
            sVtp[(cp0 + 4 * j + i) * BN + kid] = (_Float16)vreg[j][i];
}

__global__ __launch_bounds__(256) void l2attn_wmma_kernel(
    const float* __restrict__ qkv, float* __restrict__ out)
{
    __shared__ __align__(32) _Float16 sK [2][BN * DIM];     // scaled K, [key][ch]
    __shared__ __align__(32) _Float16 sVt[2][DIM * BN];     // V transposed
    __shared__ float                  sK2[2][BN];           // ||k*scale||^2
    __shared__ __align__(32) _Float16 sP [NWAVES][16 * BN]; // per-wave P scratch

    const int tid    = threadIdx.x;
    const int wave   = tid >> 5;
    const int lane   = tid & 31;
    const int laneLo = lane & 15;
    const bool hiH   = (lane >= 16);

    const int b      = blockIdx.y;
    const int qrow0  = blockIdx.x * BM + wave * 16;
    const size_t bb  = (size_t)b * TSEQ * (3 * DIM);

    // cooperative-load thread mapping: 8 threads per key, 24 channels each
    const int kid = tid >> 3;          // 0..31
    const int cp0 = (tid & 7) * 24;    // 0..168
    const float* ldbase = qkv + bb + (size_t)kid * (3 * DIM) + DIM + cp0;

    // ---------------- Q: load 16x192 strip directly into A-fragment layout ----
    v16h qfrag[KSTEPS];
    {
        const int qrow = qrow0 + laneLo;
        const float* qp = qkv + bb + (size_t)qrow * (3 * DIM);
#pragma unroll
        for (int kk = 0; kk < KSTEPS; ++kk) {
            const int c0 = kk * 32 + (hiH ? 16 : 0);
            v4f a0 = *(const v4f*)(qp + c0);
            v4f a1 = *(const v4f*)(qp + c0 + 4);
            v4f b0 = *(const v4f*)(qp + c0 + 8);
            v4f b1 = *(const v4f*)(qp + c0 + 12);
            v16h a;
#pragma unroll
            for (int i = 0; i < 4; ++i) {
                a[i]      = (_Float16)(a0[i] * SCALE_F);
                a[i + 4]  = (_Float16)(a1[i] * SCALE_F);
                a[i + 8]  = (_Float16)(b0[i] * SCALE_F);
                a[i + 12] = (_Float16)(b1[i] * SCALE_F);
            }
            qfrag[kk] = a;
        }
    }

    // ||q*scale||^2 per row, replicated into C/D-layout element order
    float q2v[8];
    {
        float part = 0.f;
#pragma unroll
        for (int kk = 0; kk < KSTEPS; ++kk)
#pragma unroll
            for (int i = 0; i < 16; ++i) {
                float x = (float)qfrag[kk][i];
                part += x * x;
            }
        float q2row = part + __shfl_xor(part, 16, 32); // lane l: q2 of row (l&15)
#pragma unroll
        for (int i = 0; i < 8; ++i)
            q2v[i] = __shfl(q2row, i + (hiH ? 8 : 0), 32);
    }

    // softmax denominators (per-lane partials; logits in [-3,0] -> no max trick)
    float lrow[8];
    v8f oacc[CT];
#pragma unroll
    for (int i = 0; i < 8; ++i) lrow[i] = 0.f;
#pragma unroll
    for (int c = 0; c < CT; ++c) oacc[c] = v8f{};

    // ---------------- software-pipelined stream over key/value tiles ----------
    v4f kreg[6], vreg[6];
    load_tile_regs(ldbase, kreg, vreg);
    store_tile_lds(sK[0], sVt[0], sK2[0], kreg, vreg, kid, cp0, tid);
    __syncthreads();

    for (int it = 0; it < NT; ++it) {
        const int p = it & 1;

        // stage tile it+1 into registers; WMMA stream below hides the latency
        if (it + 1 < NT)
            load_tile_regs(ldbase + (size_t)(it + 1) * (BN * 3 * DIM), kreg, vreg);

        // ---- S = (qK^T) for this wave's 16 rows x 32 keys --------------------
        v8f acc0 = v8f{};
        v8f acc1 = v8f{};
#pragma unroll
        for (int kk = 0; kk < KSTEPS; ++kk) {
            const int c0 = kk * 32 + (hiH ? 16 : 0);
            v16h bf0 = *(const v16h*)&sK[p][(laneLo)      * DIM + c0];
            v16h bf1 = *(const v16h*)&sK[p][(laneLo + 16) * DIM + c0];
            acc0 = __builtin_amdgcn_wmma_f32_16x16x32_f16(
                       false, qfrag[kk], false, bf0, (short)0, acc0, false, false);
            acc1 = __builtin_amdgcn_wmma_f32_16x16x32_f16(
                       false, qfrag[kk], false, bf1, (short)0, acc1, false, false);
        }

        // ---- P = exp(-dist); accumulate per-lane denominators ----------------
        const float k2a = sK2[p][laneLo];
        const float k2b = sK2[p][laneLo + 16];
        float p0[8], p1[8];
#pragma unroll
        for (int i = 0; i < 8; ++i) {
            float d0 = fmaxf(q2v[i] + k2a - 2.f * acc0[i], 0.f);
            float d1 = fmaxf(q2v[i] + k2b - 2.f * acc1[i], 0.f);
            float e0 = __expf(-__builtin_amdgcn_sqrtf(d0));
            float e1 = __expf(-__builtin_amdgcn_sqrtf(d1));
            lrow[i] += e0 + e1;
            p0[i] = e0;
            p1[i] = e1;
        }

        // ---- P: C/D layout -> A layout via per-wave LDS scratch --------------
        _Float16* pw = &sP[wave][0];
#pragma unroll
        for (int i = 0; i < 8; ++i) {
            int row = i + (hiH ? 8 : 0);
            pw[row * BN + laneLo]      = (_Float16)p0[i];
            pw[row * BN + 16 + laneLo] = (_Float16)p1[i];
        }
        v16h pa;
        {
            const int pr  = laneLo;
            const int pc0 = hiH ? 16 : 0;
            v8h plo = *(const v8h*)&pw[pr * BN + pc0];
            v8h phi = *(const v8h*)&pw[pr * BN + pc0 + 8];
#pragma unroll
            for (int i = 0; i < 8; ++i) { pa[i] = plo[i]; pa[i + 8] = phi[i]; }
        }

        // ---- O += P x V (K = 32 keys, N = 192 channels in 12 tiles) ----------
#pragma unroll
        for (int ct = 0; ct < CT; ++ct) {
            const int ch = ct * 16 + laneLo;
            v16h vbf = *(const v16h*)&sVt[p][ch * BN + (hiH ? 16 : 0)];
            oacc[ct] = __builtin_amdgcn_wmma_f32_16x16x32_f16(
                           false, pa, false, vbf, (short)0, oacc[ct], false, false);
        }

        // ---- drain staged tile into the other LDS buffer ---------------------
        if (it + 1 < NT)
            store_tile_lds(sK[p ^ 1], sVt[p ^ 1], sK2[p ^ 1],
                           kreg, vreg, kid, cp0, tid);
        __syncthreads();
    }

    // ---------------- epilogue: reduce denominators, normalize, store ---------
    float linv[8];
#pragma unroll
    for (int i = 0; i < 8; ++i) {
        float l = lrow[i];
        l += __shfl_xor(l, 1, 32);
        l += __shfl_xor(l, 2, 32);
        l += __shfl_xor(l, 4, 32);
        l += __shfl_xor(l, 8, 32);
        linv[i] = 1.f / l;             // l >= 1 (diagonal term)
    }
#pragma unroll
    for (int ct = 0; ct < CT; ++ct) {
#pragma unroll
        for (int i = 0; i < 8; ++i) {
            int row = qrow0 + i + (hiH ? 8 : 0);
            out[(size_t)(b * TSEQ + row) * DIM + ct * 16 + laneLo] =
                oacc[ct][i] * linv[i];
        }
    }
}

extern "C" void kernel_launch(void* const* d_in, const int* in_sizes, int n_in,
                              void* d_out, int out_size, void* d_ws, size_t ws_size,
                              hipStream_t stream) {
    (void)in_sizes; (void)n_in; (void)d_ws; (void)ws_size; (void)out_size;
    const float* qkv = (const float*)d_in[0];
    float* out = (float*)d_out;
    dim3 grid(TSEQ / BM, NBATCH);
    l2attn_wmma_kernel<<<grid, 256, 0, stream>>>(qkv, out);
}
